// ResidualDMPNN_326417514979
// MI455X (gfx1250) — matrix operand order repaired and checked by reading, not compiled, not run on hardware
//
#include <hip/hip_runtime.h>
#include <hip/hip_bf16.h>

// ---------------------------------------------------------------------------
// ResidualDMPNN for MI455X (gfx1250, wave32, WMMA).
// Edge GEMMs on v_wmma_f32_16x16x32_bf16; weights staged in LDS per
// workgroup; A fragments preloaded; B fragments software-pipelined (2-deep)
// from LDS; sched_group_barrier enforces DS-read / WMMA interleaving.
// ---------------------------------------------------------------------------

typedef __attribute__((ext_vector_type(16))) __bf16 v16bf;
typedef __attribute__((ext_vector_type(8)))  __bf16 v8bf;
typedef __attribute__((ext_vector_type(8)))  float  v8f;

#define AE    128
#define BE    64
#define MSGD  128
#define FEATD 192   // AE + BE
#define KFULL 320   // AE + BE + MSGD
#define OHD   256

#define WPW   8     // waves per workgroup
#define TPW   4     // 16-edge tiles per wave
#define LDSTR_I (FEATD + 8)   // LDS row stride (bf16) init:  200 -> conflict-free
#define LDSTR_B (KFULL + 8)   // LDS row stride (bf16) block: 328 -> conflict-free

// sched_group_barrier masks
#define SGB_WMMA   0x008
#define SGB_DSREAD 0x100

__device__ __forceinline__ unsigned short f2bfu(float f) {
  unsigned u = __builtin_bit_cast(unsigned, f);
  u += 0x7FFFu + ((u >> 16) & 1u);        // round-to-nearest-even
  return (unsigned short)(u >> 16);
}
__device__ __forceinline__ __bf16 f2bf(float f) {
  unsigned short s = f2bfu(f);
  return __builtin_bit_cast(__bf16, s);
}

__device__ __forceinline__ v16bf zero16bf() {
  v16bf z;
#pragma unroll
  for (int i = 0; i < 16; ++i) z[i] = __builtin_bit_cast(__bf16, (unsigned short)0);
  return z;
}

// Pipeline shape for one N-tile-pair: 8 DS reads up front, then
// {2 WMMA, 4 DS reads} x (KS-2), then 4 WMMA to drain.
template <int KS>
__device__ __forceinline__ void sched_pipeline_pair() {
  __builtin_amdgcn_sched_group_barrier(SGB_DSREAD, 8, 0);
#pragma unroll
  for (int i = 0; i < KS - 2; ++i) {
    __builtin_amdgcn_sched_group_barrier(SGB_WMMA, 2, 0);
    __builtin_amdgcn_sched_group_barrier(SGB_DSREAD, 4, 0);
  }
  __builtin_amdgcn_sched_group_barrier(SGB_WMMA, 4, 0);
}

// ---------------------------------------------------------------------------
// f32 -> bf16 weight conversion
// ---------------------------------------------------------------------------
__global__ void cvt_f32_bf16_kernel(const float* __restrict__ s,
                                    __bf16* __restrict__ d, int n) {
  int i = blockIdx.x * blockDim.x + threadIdx.x;
  if (i < n) d[i] = f2bf(s[i]);
}

// ---------------------------------------------------------------------------
// Build per-edge bf16 feature rows: feat[e] = [atom_table[x[src[e]]] || bond]
// ---------------------------------------------------------------------------
__global__ void build_edge_feat_kernel(const float* __restrict__ atom_table,
                                       const float* __restrict__ bond_table,
                                       const int* __restrict__ x,
                                       const int* __restrict__ eattr,
                                       const int* __restrict__ src,
                                       __bf16* __restrict__ feat, int E) {
  int idx = blockIdx.x * blockDim.x + threadIdx.x;
  int e = idx >> 5;
  if (e >= E) return;
  int lane = idx & 31;
  int a_row = x[src[e]];
  int b_row = eattr[e];
  const float* ap = atom_table + (size_t)a_row * AE;
  const float* bp = bond_table + (size_t)b_row * BE;
  __bf16* out = feat + (size_t)e * FEATD;
#pragma unroll
  for (int j = 0; j < 6; ++j) {
    int c = lane * 6 + j;
    float v = (c < AE) ? ap[c] : bp[c - AE];
    out[c] = f2bf(v);
  }
}

// ---------------------------------------------------------------------------
// Zero f32 buffer
// ---------------------------------------------------------------------------
__global__ void zero_f32_kernel(float* __restrict__ p, long long n) {
  long long i = (long long)blockIdx.x * blockDim.x + threadIdx.x;
  if (i < n) p[i] = 0.0f;
}

// ---------------------------------------------------------------------------
// segment_sum over edges: out[dst[e], :] += msg[e, :]   (f32 atomics at L2)
// ---------------------------------------------------------------------------
__global__ void scatter_edges_kernel(const float* __restrict__ msg,
                                     const int* __restrict__ dst,
                                     float* __restrict__ out, int E) {
  int idx = blockIdx.x * blockDim.x + threadIdx.x;
  int e = idx >> 5;
  if (e >= E) return;
  int lane = idx & 31;
  float4 v = *(const float4*)(msg + (size_t)e * MSGD + lane * 4);
  float* o = out + (size_t)dst[e] * MSGD + lane * 4;
  atomicAdd(o + 0, v.x);
  atomicAdd(o + 1, v.y);
  atomicAdd(o + 2, v.z);
  atomicAdd(o + 3, v.w);
}

// ---------------------------------------------------------------------------
// segment_sum over nodes: mol[batch[n], :] += node[n, :]
// ---------------------------------------------------------------------------
__global__ void scatter_nodes_kernel(const float* __restrict__ node,
                                     const int* __restrict__ batch,
                                     float* __restrict__ mol, int N) {
  int idx = blockIdx.x * blockDim.x + threadIdx.x;
  int n = idx >> 5;
  if (n >= N) return;
  int lane = idx & 31;
  float4 v = *(const float4*)(node + (size_t)n * MSGD + lane * 4);
  float* o = mol + (size_t)batch[n] * MSGD + lane * 4;
  atomicAdd(o + 0, v.x);
  atomicAdd(o + 1, v.y);
  atomicAdd(o + 2, v.z);
  atomicAdd(o + 3, v.w);
}

// ---------------------------------------------------------------------------
// A-fragment load (contiguous k order, same as B): lane = (m, c); row m, K-halves by c.
// ---------------------------------------------------------------------------
__device__ __forceinline__ v16bf loadA(const __bf16* arow, int k0, int c) {
  v8bf lo = *(const v8bf*)(arow + k0 + c * 16);
  v8bf hi = *(const v8bf*)(arow + k0 + c * 16 + 8);
  v16bf a;
#pragma unroll
  for (int i = 0; i < 8; ++i) { a[i] = lo[i]; a[8 + i] = hi[i]; }
  return a;
}

// ---------------------------------------------------------------------------
// Init GEMM: msg = relu(feat[E,192] @ Wb[128,192]^T + bias)
// ---------------------------------------------------------------------------
__global__ __launch_bounds__(WPW * 32, 1)
void gemm_init_kernel(const __bf16* __restrict__ feat,
                      const __bf16* __restrict__ Wb,
                      const float* __restrict__ bias,
                      float* __restrict__ msg, int numTiles) {
  __shared__ __bf16 Ws[128 * LDSTR_I];   // 128 x 200 bf16 = 50 KB
  constexpr int KS = FEATD / 32;         // 6

  for (int i = threadIdx.x; i < 128 * (FEATD / 8); i += WPW * 32) {
    int n = i / (FEATD / 8), j = i % (FEATD / 8);
    *(uint4*)&Ws[n * LDSTR_I + j * 8] =
        *(const uint4*)(Wb + (size_t)n * FEATD + j * 8);
  }
  __syncthreads();

  const int wave = threadIdx.x >> 5;
  const int l    = threadIdx.x & 31;
  const int m    = l & 15;
  const int c    = l >> 4;
  const int kb0  = c * 16;

  for (int it = 0; it < TPW; ++it) {
    const int tile = (blockIdx.x * WPW + wave) * TPW + it;
    if (tile >= numTiles) break;
    const int e0 = tile * 16;
    const __bf16* arow = feat + (size_t)(e0 + m) * FEATD;
    __builtin_prefetch(arow + 16 * FEATD, 0, 1);

    // preload all A fragments (per-wave private, pay global latency once)
    v16bf af[KS];
#pragma unroll
    for (int ks = 0; ks < KS; ++ks) af[ks] = loadA(arow, ks * 32, c);

    v8f upd[8];
    // N-tiles in pairs: two independent accumulate chains, B pipelined 2-deep
#pragma unroll
    for (int tp = 0; tp < 4; ++tp) {
      const __bf16* brow0 = &Ws[(tp * 32 + m) * LDSTR_I + kb0];
      const __bf16* brow1 = &Ws[(tp * 32 + 16 + m) * LDSTR_I + kb0];
      v16bf b0[KS], b1[KS];
      b0[0] = *(const v16bf*)(brow0);
      b1[0] = *(const v16bf*)(brow1);
      b0[1] = *(const v16bf*)(brow0 + 32);
      b1[1] = *(const v16bf*)(brow1 + 32);
      v8f a0, a1;
#pragma unroll
      for (int i = 0; i < 8; ++i) { a0[i] = 0.0f; a1[i] = 0.0f; }
#pragma unroll
      for (int ks = 0; ks < KS; ++ks) {
        if (ks + 2 < KS) {
          b0[ks + 2] = *(const v16bf*)(brow0 + (ks + 2) * 32);
          b1[ks + 2] = *(const v16bf*)(brow1 + (ks + 2) * 32);
        }
        a0 = __builtin_amdgcn_wmma_f32_16x16x32_bf16(
            false, af[ks], false, b0[ks], (short)0, a0, false, false);
        a1 = __builtin_amdgcn_wmma_f32_16x16x32_bf16(
            false, af[ks], false, b1[ks], (short)0, a1, false, false);
      }
      sched_pipeline_pair<KS>();
      upd[tp * 2]     = a0;
      upd[tp * 2 + 1] = a1;
    }

#pragma unroll
    for (int t = 0; t < 8; ++t) {
      const int n  = t * 16 + m;
      const float bv = bias[n];
#pragma unroll
      for (int v = 0; v < 8; ++v) {
        float y = upd[t][v] + bv;
        y = y > 0.0f ? y : 0.0f;
        msg[(size_t)(e0 + v + 8 * c) * MSGD + n] = y;
      }
    }
  }
}

// ---------------------------------------------------------------------------
// Residual block GEMM + residual + LayerNorm (epilogue in registers):
//   upd = relu([feat || inc[src]] @ Wl^T + b);  msg = LN(msg + upd)*g + beta
// ---------------------------------------------------------------------------
__global__ __launch_bounds__(WPW * 32, 1)
void gemm_block_kernel(const __bf16* __restrict__ feat,
                       const float* __restrict__ inc,
                       const int* __restrict__ src,
                       const __bf16* __restrict__ Wb,
                       const float* __restrict__ bias,
                       const float* __restrict__ gam,
                       const float* __restrict__ beta,
                       float* __restrict__ msg, int numTiles) {
  __shared__ __bf16 Ws[128 * LDSTR_B];   // 128 x 328 bf16 = 82 KB
  constexpr int KS = KFULL / 32;         // 10

  for (int i = threadIdx.x; i < 128 * (KFULL / 8); i += WPW * 32) {
    int n = i / (KFULL / 8), j = i % (KFULL / 8);
    *(uint4*)&Ws[n * LDSTR_B + j * 8] =
        *(const uint4*)(Wb + (size_t)n * KFULL + j * 8);
  }
  __syncthreads();

  const int wave = threadIdx.x >> 5;
  const int l    = threadIdx.x & 31;
  const int m    = l & 15;
  const int c    = l >> 4;
  const int kb0  = c * 16;

  for (int it = 0; it < TPW; ++it) {
    const int tile = (blockIdx.x * WPW + wave) * TPW + it;
    if (tile >= numTiles) break;
    const int e0 = tile * 16;
    const __bf16* arow = feat + (size_t)(e0 + m) * FEATD;
    const float*  irow = inc + (size_t)src[e0 + m] * MSGD;
    __builtin_prefetch(arow + 16 * FEATD, 0, 1);

    // preload all 10 A fragments: 6 from bf16 feat, 4 from f32 inc gather
    v16bf af[KS];
#pragma unroll
    for (int ks = 0; ks < FEATD / 32; ++ks) af[ks] = loadA(arow, ks * 32, c);
#pragma unroll
    for (int ks = FEATD / 32; ks < KS; ++ks) {
      const float* p = irow + (ks * 32 - FEATD) + c * 16;
      float4 f0 = *(const float4*)(p);
      float4 f1 = *(const float4*)(p + 4);
      float4 f2 = *(const float4*)(p + 8);
      float4 f3 = *(const float4*)(p + 12);
      v16bf a = zero16bf();
      a[0]  = f2bf(f0.x); a[1]  = f2bf(f0.y); a[2]  = f2bf(f0.z); a[3]  = f2bf(f0.w);
      a[4]  = f2bf(f1.x); a[5]  = f2bf(f1.y); a[6]  = f2bf(f1.z); a[7]  = f2bf(f1.w);
      a[8]  = f2bf(f2.x); a[9]  = f2bf(f2.y); a[10] = f2bf(f2.z); a[11] = f2bf(f2.w);
      a[12] = f2bf(f3.x); a[13] = f2bf(f3.y); a[14] = f2bf(f3.z); a[15] = f2bf(f3.w);
      af[ks] = a;
    }

    v8f upd[8];
#pragma unroll
    for (int tp = 0; tp < 4; ++tp) {
      const __bf16* brow0 = &Ws[(tp * 32 + m) * LDSTR_B + kb0];
      const __bf16* brow1 = &Ws[(tp * 32 + 16 + m) * LDSTR_B + kb0];
      v16bf b0[KS], b1[KS];
      b0[0] = *(const v16bf*)(brow0);
      b1[0] = *(const v16bf*)(brow1);
      b0[1] = *(const v16bf*)(brow0 + 32);
      b1[1] = *(const v16bf*)(brow1 + 32);
      v8f a0, a1;
#pragma unroll
      for (int i = 0; i < 8; ++i) { a0[i] = 0.0f; a1[i] = 0.0f; }
#pragma unroll
      for (int ks = 0; ks < KS; ++ks) {
        if (ks + 2 < KS) {
          b0[ks + 2] = *(const v16bf*)(brow0 + (ks + 2) * 32);
          b1[ks + 2] = *(const v16bf*)(brow1 + (ks + 2) * 32);
        }
        a0 = __builtin_amdgcn_wmma_f32_16x16x32_bf16(
            false, af[ks], false, b0[ks], (short)0, a0, false, false);
        a1 = __builtin_amdgcn_wmma_f32_16x16x32_bf16(
            false, af[ks], false, b1[ks], (short)0, a1, false, false);
      }
      sched_pipeline_pair<KS>();
      upd[tp * 2]     = a0;
      upd[tp * 2 + 1] = a1;
    }

    // ---- epilogue in registers: bias + relu + residual ----
#pragma unroll
    for (int t = 0; t < 8; ++t) {
      const int n  = t * 16 + m;
      const float bv = bias[n];
#pragma unroll
      for (int v = 0; v < 8; ++v) {
        float y = upd[t][v] + bv;
        y = y > 0.0f ? y : 0.0f;
        upd[t][v] = y + msg[(size_t)(e0 + v + 8 * c) * MSGD + n];
      }
    }
    // row stats over 128 channels: in-lane over t, then 16-lane shfl_xor
    float mu[8], rstd[8];
#pragma unroll
    for (int v = 0; v < 8; ++v) {
      float s = 0.0f;
#pragma unroll
      for (int t = 0; t < 8; ++t) s += upd[t][v];
#pragma unroll
      for (int off = 1; off < 16; off <<= 1) s += __shfl_xor(s, off, 32);
      mu[v] = s * (1.0f / 128.0f);
    }
#pragma unroll
    for (int v = 0; v < 8; ++v) {
      float q = 0.0f;
#pragma unroll
      for (int t = 0; t < 8; ++t) {
        float d = upd[t][v] - mu[v];
        q += d * d;
      }
#pragma unroll
      for (int off = 1; off < 16; off <<= 1) q += __shfl_xor(q, off, 32);
      rstd[v] = rsqrtf(q * (1.0f / 128.0f) + 1e-5f);
    }
#pragma unroll
    for (int t = 0; t < 8; ++t) {
      const int n  = t * 16 + m;
      const float gv = gam[n];
      const float tv = beta[n];
#pragma unroll
      for (int v = 0; v < 8; ++v) {
        msg[(size_t)(e0 + v + 8 * c) * MSGD + n] =
            (upd[t][v] - mu[v]) * rstd[v] * gv + tv;
      }
    }
  }
}

// ---------------------------------------------------------------------------
// Readout MLP: out[g] = relu(mol[g] @ w1^T + b1) @ w2^T + b2
// ---------------------------------------------------------------------------
__global__ __launch_bounds__(256)
void readout_kernel(const float* __restrict__ mol,
                    const float* __restrict__ w1, const float* __restrict__ b1,
                    const float* __restrict__ w2, const float* __restrict__ b2,
                    float* __restrict__ out, int G) {
  __shared__ float red[OHD];
  const int g = blockIdx.x;
  const int o = threadIdx.x;
  const float* mrow = mol + (size_t)g * MSGD;
  const float* wrow = w1 + (size_t)o * MSGD;
  float s = b1[o];
#pragma unroll 8
  for (int k = 0; k < MSGD; ++k) s += mrow[k] * wrow[k];
  s = s > 0.0f ? s : 0.0f;
  red[o] = s * w2[o];
  __syncthreads();
  for (int st = OHD / 2; st > 0; st >>= 1) {
    if (o < st) red[o] += red[o + st];
    __syncthreads();
  }
  if (o == 0) out[g] = red[0] + b2[0];
}

// ---------------------------------------------------------------------------
// Host orchestration
// ---------------------------------------------------------------------------
extern "C" void kernel_launch(void* const* d_in, const int* in_sizes, int n_in,
                              void* d_out, int out_size, void* d_ws, size_t ws_size,
                              hipStream_t stream) {
  const float* atom_table = (const float*)d_in[0];
  const float* bond_table = (const float*)d_in[1];
  const float* w_init     = (const float*)d_in[2];
  const float* b_init     = (const float*)d_in[3];
  const float* blk_w      = (const float*)d_in[4];
  const float* blk_b      = (const float*)d_in[5];
  const float* blk_g      = (const float*)d_in[6];
  const float* blk_beta   = (const float*)d_in[7];
  const float* w1         = (const float*)d_in[8];
  const float* b1         = (const float*)d_in[9];
  const float* w2         = (const float*)d_in[10];
  const float* b2         = (const float*)d_in[11];
  const int*   x          = (const int*)d_in[12];
  const int*   eattr      = (const int*)d_in[13];
  const int*   eidx       = (const int*)d_in[14];
  const int*   batch      = (const int*)d_in[15];

  const int N = in_sizes[12];
  const int E = in_sizes[13];
  const int G = out_size;
  const int* src = eidx;
  const int* dst = eidx + E;

  // workspace carve-out
  char* ws = (char*)d_ws;
  size_t off = 0;
  auto carve = [&](size_t bytes) -> void* {
    void* p = ws + off;
    off = (off + bytes + 255) & ~(size_t)255;
    return p;
  };
  __bf16* wInitB = (__bf16*)carve((size_t)MSGD * FEATD * 2);      //  48 KB
  __bf16* blkWB  = (__bf16*)carve((size_t)4 * MSGD * KFULL * 2);  // 320 KB
  __bf16* featB  = (__bf16*)carve((size_t)E * FEATD * 2);         // 192 MB
  float*  msg    = (float*)carve((size_t)E * MSGD * 4);           // 256 MB
  float*  inc    = (float*)carve((size_t)N * MSGD * 4);           //  26 MB
  float*  mol    = (float*)carve((size_t)G * MSGD * 4);           //   1 MB
  (void)ws_size;

  const int T = 256;
  const int numTiles = E / 16;                       // E % 16 == 0 (500000/16)
  const int gemmWGs = (numTiles + WPW * TPW - 1) / (WPW * TPW);

  // weights -> bf16
  {
    int n0 = MSGD * FEATD;
    int n1 = 4 * MSGD * KFULL;
    cvt_f32_bf16_kernel<<<(n0 + T - 1) / T, T, 0, stream>>>(w_init, wInitB, n0);
    cvt_f32_bf16_kernel<<<(n1 + T - 1) / T, T, 0, stream>>>(blk_w, blkWB, n1);
  }
  // per-edge [atom_src || bond] bf16 features
  {
    long long nt = (long long)E * 32;
    build_edge_feat_kernel<<<(unsigned)((nt + T - 1) / T), T, 0, stream>>>(
        atom_table, bond_table, x, eattr, src, featB, E);
  }
  // msg init GEMM (WMMA)
  gemm_init_kernel<<<gemmWGs, WPW * 32, 0, stream>>>(featB, wInitB, b_init, msg,
                                                     numTiles);

  // 4 residual message-passing blocks
  for (int l = 0; l < 4; ++l) {
    long long ni = (long long)N * MSGD;
    zero_f32_kernel<<<(unsigned)((ni + T - 1) / T), T, 0, stream>>>(inc, ni);
    long long nt = (long long)E * 32;
    scatter_edges_kernel<<<(unsigned)((nt + T - 1) / T), T, 0, stream>>>(msg, dst, inc, E);
    gemm_block_kernel<<<gemmWGs, WPW * 32, 0, stream>>>(
        featB, inc, src,
        blkWB + (size_t)l * MSGD * KFULL,
        blk_b + (size_t)l * MSGD,
        blk_g + (size_t)l * MSGD,
        blk_beta + (size_t)l * MSGD,
        msg, numTiles);
  }

  // node_state = segment_sum(msg, dst); mol_state = segment_sum(node, batch)
  {
    long long ni = (long long)N * MSGD;
    zero_f32_kernel<<<(unsigned)((ni + T - 1) / T), T, 0, stream>>>(inc, ni);
    long long nt = (long long)E * 32;
    scatter_edges_kernel<<<(unsigned)((nt + T - 1) / T), T, 0, stream>>>(msg, dst, inc, E);
    long long nm = (long long)G * MSGD;
    zero_f32_kernel<<<(unsigned)((nm + T - 1) / T), T, 0, stream>>>(mol, nm);
    long long nn = (long long)N * 32;
    scatter_nodes_kernel<<<(unsigned)((nn + T - 1) / T), T, 0, stream>>>(inc, batch, mol, N);
  }

  // final MLP readout
  readout_kernel<<<G, OHD, 0, stream>>>(mol, w1, b1, w2, b2, (float*)d_out, G);
}
